// Mult_CA_v1_60885456388400
// MI455X (gfx1250) — matrix operand order repaired and checked by reading, not compiled, not run on hardware
//
#include <hip/hip_runtime.h>
#include <hip/hip_bf16.h>
#include <stdint.h>

#define N_TOK  32768
#define EMBED  512
#define N_CLS  16      // L
#define BATCH  8
#define N_HEAD 8
#define D_HEAD 64
#define TILE_M 32                  // tokens per block (2 WMMA M-subtiles)
#define EPAIRS (EMBED/2)           // 256 dwords (bf16 pairs) per 512-wide row
#define KSTEPS (EMBED/32)          // 16 k-steps per GEMM
#define FTILES (EMBED/16)          // 32 output-feature tiles

typedef __attribute__((ext_vector_type(16))) __bf16 v16bf;
typedef __attribute__((ext_vector_type(8)))  float  v8f;

__device__ __forceinline__ uint16_t f32_to_bf16(float f) {
    uint32_t u = __float_as_uint(f);
    uint32_t r = u + 0x7FFFu + ((u >> 16) & 1u);   // round-to-nearest-even
    return (uint16_t)(r >> 16);
}
__device__ __forceinline__ uint32_t pack_bf16x2(float lo, float hi) {
    return (uint32_t)f32_to_bf16(lo) | ((uint32_t)f32_to_bf16(hi) << 16);
}

// ---- prep 1: fp32 weight -> bf16, pre-swizzled into WMMA B-fragment order ---
// out dword index i = ((ftile*KSTEPS + kstep)*32 + lane)*8 + j
//   column n = lane&15  -> weight row f = ftile*16 + n
//   kg = lane>>4        -> K element = kstep*32 + (j&4)*4 + kg*8 + 2(j&3) (+1)
// so a wave's B fragment for (ftile,kstep) is one contiguous 1KB block.
__global__ __launch_bounds__(256) void convert_w_kernel(
    const float* __restrict__ W, uint32_t* __restrict__ Wb, int n_dwords)
{
    int i = blockIdx.x * 256 + threadIdx.x;
    if (i >= n_dwords) return;
    int j     = i & 7;
    int lane  = (i >> 3) & 31;
    int kstep = (i >> 8) & (KSTEPS - 1);
    int ftile = i >> 12;
    int f  = ftile * 16 + (lane & 15);
    int kg = lane >> 4;
    size_t srcpair = (size_t)f * EPAIRS + kstep * 16 + (j & 4) * 2 + kg * 4 + (j & 3);
    Wb[i] = pack_bf16x2(W[2 * srcpair], W[2 * srcpair + 1]);
}

// ---- prep 2: k/v projections, fp32, layout [B][L][EMBED] --------------------
__global__ __launch_bounds__(256) void kv_proj_kernel(
    const float* __restrict__ key, const float* __restrict__ val,
    const float* __restrict__ Wk,  const float* __restrict__ Wv,
    const float* __restrict__ bk,  const float* __restrict__ bv,
    float* __restrict__ kp, float* __restrict__ vp)
{
    int o = blockIdx.x * 256 + threadIdx.x;        // 0 .. BATCH*N_CLS*EMBED-1
    int f = o & (EMBED - 1);
    int l = (o >> 9) & (N_CLS - 1);
    int b = o >> 13;
    const float* krow = key + ((size_t)l * BATCH + b) * EMBED;  // key[l][b][:]
    const float* vrow = val + ((size_t)l * BATCH + b) * EMBED;
    const float* wkr  = Wk  + (size_t)f * EMBED;
    const float* wvr  = Wv  + (size_t)f * EMBED;
    float sk = 0.f, sv = 0.f;
    #pragma unroll 4
    for (int e = 0; e < EMBED; ++e) { sk += krow[e] * wkr[e]; sv += vrow[e] * wvr[e]; }
    kp[o] = sk + bk[f];
    vp[o] = sv + bv[f];
}

// ---- fused main kernel ------------------------------------------------------
__global__ __launch_bounds__(256) void fused_attn_kernel(
    const float*    __restrict__ query,
    const int*      __restrict__ batch_idx,
    const uint32_t* __restrict__ WqB,   // swizzled bf16 fragments
    const uint32_t* __restrict__ WoB,
    const float*    __restrict__ bq,
    const float*    __restrict__ bo,
    const float*    __restrict__ kp,    // [B][L][512] fp32
    const float*    __restrict__ vp,
    float*          __restrict__ out)
{
    // sbuf: query tile (bf16 pairs) for GEMM1, then reused as ctx tile for GEMM2
    __shared__ uint32_t sbuf[TILE_M][EPAIRS + 4];
    __shared__ float    sQ[TILE_M][EMBED + 4];     // q-projection, fp32

    const int tid  = threadIdx.x;
    const int wave = tid >> 5;
    const int lane = tid & 31;
    const int row  = lane & 15;     // A-row-within-subtile / C-column
    const int kg   = lane >> 4;     // K-group select per ISA layout
    const int n0   = blockIdx.x * TILE_M;

    // ---- stage query tile -> bf16 in LDS (32 rows x 256 dwords)
    for (int i = tid; i < TILE_M * EPAIRS; i += 256) {
        int r = i >> 8, c = i & (EPAIRS - 1);
        const float* qrow = query + (size_t)(n0 + r) * EMBED + 2 * c;
        sbuf[r][c] = pack_bf16x2(qrow[0], qrow[1]);
    }
    __syncthreads();

    // ---- GEMM1: q = query @ Wq^T  (wave: 4 f-tiles x 2 M-subtiles)
    #pragma unroll
    for (int t = 0; t < 4; ++t) {
        const int ft = wave * 4 + t;
        const int f0 = ft * 16;
        v8f acc0 = {}, acc1 = {};
        for (int ks = 0; ks < KSTEPS; ++ks) {
            union { uint32_t u[8]; v16bf v; } a0, a1, b;
            const uint32_t* bp = WqB + (((size_t)ft * KSTEPS + ks) * 32 + lane) * 8;
            #pragma unroll
            for (int j = 0; j < 8; ++j) b.u[j] = bp[j];
            if (ks + 1 < KSTEPS) __builtin_prefetch(bp + 32 * 8, 0, 1);
            const int abase = (ks * 32 + kg * 8) >> 1;     // dword index
            #pragma unroll
            for (int j = 0; j < 4; ++j) {
                a0.u[j]     = sbuf[row][abase + j];
                a0.u[4 + j] = sbuf[row][abase + 8 + j];
                a1.u[j]     = sbuf[16 + row][abase + j];
                a1.u[4 + j] = sbuf[16 + row][abase + 8 + j];
            }
            acc0 = __builtin_amdgcn_wmma_f32_16x16x32_bf16(false, a0.v, false, b.v,
                                                           (short)0, acc0, false, false);
            acc1 = __builtin_amdgcn_wmma_f32_16x16x32_bf16(false, a1.v, false, b.v,
                                                           (short)0, acc1, false, false);
        }
        const float bias = bq[f0 + row];
        const int m0 = kg * 8;
        #pragma unroll
        for (int r = 0; r < 8; ++r) {
            sQ[m0 + r][f0 + row]      = acc0[r] + bias;
            sQ[16 + m0 + r][f0 + row] = acc1[r] + bias;
        }
    }
    __syncthreads();

    // ---- attention: 256 threads, one per (token, head); k/v are L2-hot
    {
        const int t = tid >> 3, h = tid & 7;          // t: 0..31, h: 0..7
        const int b = batch_idx[n0 + t];
        const float* qv    = &sQ[t][h * D_HEAD];
        const float* kbase = kp + ((size_t)b * N_CLS) * EMBED + h * D_HEAD;
        const float* vbase = vp + ((size_t)b * N_CLS) * EMBED + h * D_HEAD;
        float sc[N_CLS];
        float mx = -1e30f;
        #pragma unroll
        for (int l = 0; l < N_CLS; ++l) {
            const float* kv = kbase + (size_t)l * EMBED;
            float s = 0.f;
            #pragma unroll 8
            for (int d = 0; d < D_HEAD; ++d) s += qv[d] * kv[d];
            s *= 0.125f;                               // D_HEAD^-0.5
            sc[l] = s;
            mx = fmaxf(mx, s);
        }
        float den = 0.f;
        #pragma unroll
        for (int l = 0; l < N_CLS; ++l) { sc[l] = __expf(sc[l] - mx); den += sc[l]; }
        const float inv = 1.f / den;
        for (int d = 0; d < D_HEAD; d += 2) {
            float c0 = 0.f, c1 = 0.f;
            #pragma unroll
            for (int l = 0; l < N_CLS; ++l) {
                const float* vv = vbase + (size_t)l * EMBED;
                const float p = sc[l] * inv;
                c0 += p * vv[d];
                c1 += p * vv[d + 1];
            }
            sbuf[t][(h * D_HEAD + d) >> 1] = pack_bf16x2(c0, c1);   // ctx tile (aliases query buf)
        }
    }
    __syncthreads();

    // ---- GEMM2: out = ctx @ Wo^T
    #pragma unroll
    for (int t = 0; t < 4; ++t) {
        const int ft = wave * 4 + t;
        const int f0 = ft * 16;
        v8f acc0 = {}, acc1 = {};
        for (int ks = 0; ks < KSTEPS; ++ks) {
            union { uint32_t u[8]; v16bf v; } a0, a1, b;
            const uint32_t* bp = WoB + (((size_t)ft * KSTEPS + ks) * 32 + lane) * 8;
            #pragma unroll
            for (int j = 0; j < 8; ++j) b.u[j] = bp[j];
            if (ks + 1 < KSTEPS) __builtin_prefetch(bp + 32 * 8, 0, 1);
            const int abase = (ks * 32 + kg * 8) >> 1;
            #pragma unroll
            for (int j = 0; j < 4; ++j) {
                a0.u[j]     = sbuf[row][abase + j];
                a0.u[4 + j] = sbuf[row][abase + 8 + j];
                a1.u[j]     = sbuf[16 + row][abase + j];
                a1.u[4 + j] = sbuf[16 + row][abase + 8 + j];
            }
            acc0 = __builtin_amdgcn_wmma_f32_16x16x32_bf16(false, a0.v, false, b.v,
                                                           (short)0, acc0, false, false);
            acc1 = __builtin_amdgcn_wmma_f32_16x16x32_bf16(false, a1.v, false, b.v,
                                                           (short)0, acc1, false, false);
        }
        const float bias = bo[f0 + row];
        const int m0 = kg * 8;
        #pragma unroll
        for (int r = 0; r < 8; ++r) {
            out[(size_t)(n0 + m0 + r) * EMBED + f0 + row]      = acc0[r] + bias;
            out[(size_t)(n0 + 16 + m0 + r) * EMBED + f0 + row] = acc1[r] + bias;
        }
    }
}

// ---- host launch ------------------------------------------------------------
extern "C" void kernel_launch(void* const* d_in, const int* in_sizes, int n_in,
                              void* d_out, int out_size, void* d_ws, size_t ws_size,
                              hipStream_t stream) {
    const float* query = (const float*)d_in[0];
    const float* key   = (const float*)d_in[1];
    const float* val   = (const float*)d_in[2];
    const int*   bidx  = (const int*)  d_in[3];
    // d_in[4] = batch_size scalar (compile-time BATCH used)
    const float* Wq = (const float*)d_in[5];
    const float* bq = (const float*)d_in[6];
    const float* Wk = (const float*)d_in[7];
    const float* bk = (const float*)d_in[8];
    const float* Wv = (const float*)d_in[9];
    const float* bv = (const float*)d_in[10];
    const float* Wo = (const float*)d_in[11];
    const float* bo = (const float*)d_in[12];
    float* out = (float*)d_out;

    uint8_t* ws = (uint8_t*)d_ws;
    const int wdwords = EMBED * EPAIRS;                      // 131072 dwords
    uint32_t* WqB = (uint32_t*)ws;                           // 512 KB
    uint32_t* WoB = (uint32_t*)(ws + (size_t)wdwords * 4);   // 512 KB
    float*    kp  = (float*)(ws + 2 * (size_t)wdwords * 4);  // 256 KB
    float*    vp  = (float*)(ws + 2 * (size_t)wdwords * 4 + (size_t)BATCH * N_CLS * EMBED * 4);

    convert_w_kernel<<<(wdwords + 255) / 256, 256, 0, stream>>>(Wq, WqB, wdwords);
    convert_w_kernel<<<(wdwords + 255) / 256, 256, 0, stream>>>(Wo, WoB, wdwords);
    kv_proj_kernel<<<(BATCH * N_CLS * EMBED) / 256, 256, 0, stream>>>(key, val, Wk, Wv, bk, bv, kp, vp);
    fused_attn_kernel<<<N_TOK / TILE_M, 256, 0, stream>>>(query, bidx, WqB, WoB, bq, bo, kp, vp, out);
}
